// HVTSurv_53601191854271
// MI455X (gfx1250) — matrix operand order repaired and checked by reading, not compiled, not run on hardware
//
#include <hip/hip_runtime.h>
#include <math.h>
#include <stdint.h>

// ---------------------------------------------------------------------------
// HVTSurv forward on MI455X (gfx1250, wave32).
// Large GEMMs run through v_wmma_f32_16x16x32_bf16 (bf16 A/B, f32 accum).
// Weights are pre-transposed to [N,K] so both A and B fragments are fed by
// contiguous global_load_b128; each wave computes a 16x32 tile (2 WMMAs per
// K-step sharing one A fragment), K-loop unrolled 4x for issue density.
// ---------------------------------------------------------------------------

typedef __bf16 bf16;
typedef __attribute__((ext_vector_type(16))) __bf16 v16bf;
typedef __attribute__((ext_vector_type(8)))  float  v8f;
typedef __attribute__((ext_vector_type(4)))  unsigned int u32v4;

#define SLIDES 4
#define NTOK   9800
#define MTOK   39200          // SLIDES * NTOK
#define WSZ    49
#define NWIN   200            // NTOK / WSZ
#define CDIM   512
#define HEADS  8
#define HDIM   64
#define FDIM   1024
#define SCALE  0.125f         // 64^-0.5

__device__ __forceinline__ float gelu_exact(float x) {
  return 0.5f * x * (1.f + erff(x * 0.70710678118654752f));
}

// ----------------------------- f32 -> bf16 convert -------------------------
__global__ void k_cvt_f32_bf16(const float* __restrict__ in, bf16* __restrict__ out, size_t n) {
  size_t i = (size_t)blockIdx.x * blockDim.x + threadIdx.x;
  if (i < n) out[i] = (bf16)in[i];
}

// f32 [K,N] row-major -> bf16 [N,K] row-major (both sides coalesced via LDS)
__global__ void k_cvt_transpose_bf16(const float* __restrict__ in, bf16* __restrict__ out,
                                     int K, int N) {
  __shared__ float tile[32][33];
  const int kt = blockIdx.x * 32, nt = blockIdx.y * 32;
  const int tx = threadIdx.x & 31, ty = threadIdx.x >> 5;   // 32 x 8
  for (int r = ty; r < 32; r += 8)
    tile[r][tx] = in[(size_t)(kt + r) * N + nt + tx];
  __syncthreads();
  for (int r = ty; r < 32; r += 8)
    out[(size_t)(nt + r) * K + kt + tx] = (bf16)tile[tx][r];
}

// ----------------------------- WMMA bf16 GEMM ------------------------------
// C[M,N] = act( res + A[M,K] @ B[K,N] + bias[N] ), B given transposed [N,K].
// One wave computes a 16x32 tile (two 16x16 WMMA column tiles sharing the A
// fragment); 8 waves per block stack along M.
// ACT: 0=none, 1=relu, 2=gelu(exact), 3=tanh
template <int ACT, bool RES>
__global__ void k_gemm_bf16_wmma(const bf16* __restrict__ A, const bf16* __restrict__ BT,
                                 const float* __restrict__ bias, const float* __restrict__ res,
                                 float* __restrict__ C, int Mg, int Ng, int Kg) {
  const int lane = threadIdx.x & 31;
  const int wave = threadIdx.x >> 5;
  const int mt   = blockIdx.y * 8 + wave;      // wave-uniform
  if (mt * 16 >= Mg) return;                   // whole-wave exit keeps EXEC all-1s
  const int m0   = mt * 16;
  const int n0   = blockIdx.x * 32;
  const int kb   = lane >> 4;                  // lane half (K sub-block select)
  const int lrow = lane & 15;
  const int col0 = n0 + lrow;
  const int col1 = col0 + 16;

  // A 16x32 fragment: lane holds row m0+lrow; VGPR v -> K=kb*16+2v{+1}
  //  => one contiguous 16-half run at +kb*16 (same K order as B).
  const bf16* Arow = A  + (size_t)(m0 + lrow) * Kg + kb * 16;
  // B 32x16 fragment: lane holds column col; VGPR v -> K=kb*16+2v{+1}
  //  => one contiguous 16-half run at +kb*16 in the transposed row.
  const bf16* B0   = BT + (size_t)col0 * Kg + kb * 16;
  const bf16* B1   = BT + (size_t)col1 * Kg + kb * 16;

  v8f acc0 = {}, acc1 = {};

#pragma unroll 4
  for (int k0 = 0; k0 < Kg; k0 += 32) {
    v16bf a, b0, b1;
    ((u32v4*)&a)[0]  = *(const u32v4*)(Arow + k0);
    ((u32v4*)&a)[1]  = *(const u32v4*)(Arow + k0 + 8);
    ((u32v4*)&b0)[0] = *(const u32v4*)(B0 + k0);
    ((u32v4*)&b0)[1] = *(const u32v4*)(B0 + k0 + 8);
    ((u32v4*)&b1)[0] = *(const u32v4*)(B1 + k0);
    ((u32v4*)&b1)[1] = *(const u32v4*)(B1 + k0 + 8);
    __builtin_prefetch(Arow + k0 + 128, 0, 0);   // global_prefetch of next tiles
    __builtin_prefetch(B0 + k0 + 128, 0, 0);
    __builtin_prefetch(B1 + k0 + 128, 0, 0);
    acc0 = __builtin_amdgcn_wmma_f32_16x16x32_bf16(false, a, false, b0, (short)0, acc0,
                                                   false, false);
    acc1 = __builtin_amdgcn_wmma_f32_16x16x32_bf16(false, a, false, b1, (short)0, acc1,
                                                   false, false);
  }

  const float bv0 = bias[col0];
  const float bv1 = bias[col1];
#pragma unroll
  for (int r = 0; r < 8; ++r) {
    int mo = m0 + (kb ? 8 : 0) + r;            // C/D layout: lanes 16-31 hold M = r + 8
    float v0 = acc0[r] + bv0;
    float v1 = acc1[r] + bv1;
    if (RES) {
      v0 += res[(size_t)mo * Ng + col0];
      v1 += res[(size_t)mo * Ng + col1];
    }
    if (ACT == 1)      { v0 = fmaxf(v0, 0.f);  v1 = fmaxf(v1, 0.f); }
    else if (ACT == 2) { v0 = gelu_exact(v0);  v1 = gelu_exact(v1); }
    else if (ACT == 3) { v0 = tanhf(v0);       v1 = tanhf(v1); }
    C[(size_t)mo * Ng + col0] = v0;
    C[(size_t)mo * Ng + col1] = v1;
  }
}

// ----------------------------- LayerNorm (row = 512) -----------------------
__global__ void k_ln512(const float* __restrict__ x, const float* __restrict__ g,
                        const float* __restrict__ b, float* __restrict__ outf,
                        bf16* __restrict__ outb) {
  __shared__ float s1[256], s2[256];
  const int tid = threadIdx.x;
  const float* row = x + (size_t)blockIdx.x * CDIM;
  float a0 = row[tid], a1 = row[tid + 256];
  s1[tid] = a0 + a1;
  s2[tid] = a0 * a0 + a1 * a1;
  __syncthreads();
  for (int s = 128; s > 0; s >>= 1) {
    if (tid < s) { s1[tid] += s1[tid + s]; s2[tid] += s2[tid + s]; }
    __syncthreads();
  }
  float mean = s1[0] * (1.f / CDIM);
  float var  = s2[0] * (1.f / CDIM) - mean * mean;
  float rstd = rsqrtf(var + 1e-5f);
  float y0 = (a0 - mean) * rstd * g[tid] + b[tid];
  float y1 = (a1 - mean) * rstd * g[tid + 256] + b[tid + 256];
  size_t base = (size_t)blockIdx.x * CDIM;
  if (outf) { outf[base + tid] = y0; outf[base + tid + 256] = y1; }
  if (outb) { outb[base + tid] = (bf16)y0; outb[base + tid + 256] = (bf16)y1; }
}

// ----------------------------- windowed attention core ---------------------
__device__ __forceinline__ int pw_idx(int rp) {
  const float ALPHA = 1.9f, BETA = 7.6f, GAMMA = 11.4f;  // beta=1.9*4, gamma=1.9*6
  float rpf = (float)rp;
  float ra  = fabsf(rpf);
  float sgn = (rpf > 0.f) ? 1.f : ((rpf < 0.f) ? -1.f : 0.f);
  if (ra <= 2.f * ALPHA) return (int)sgn;
  float y = roundf(logf(ra / ALPHA) / logf(GAMMA / ALPHA) * (BETA - 2.f * ALPHA));
  y = fminf(y, 7.f);
  return (int)(sgn * y);
}

// qkv: [MTOK, 1536] (q|k|v each [H, 64] per token). out: [MTOK, 512].
// shuffle=0: window w tokens = w*49 + i, with relative-position bias from coords.
// shuffle=1: window w tokens = i*200 + w, no bias.
__global__ void k_window_attn(const float* __restrict__ qkv, const float* __restrict__ rpb,
                              const int* __restrict__ coords, float* __restrict__ out,
                              int shuffle) {
  const int w = blockIdx.x, h = blockIdx.y, bslide = blockIdx.z;
  const int tid = threadIdx.x;
  __shared__ float qs[WSZ * HDIM], ks[WSZ * HDIM], vs[WSZ * HDIM];
  __shared__ float sc[WSZ * WSZ];
  __shared__ int cx[WSZ], cy[WSZ];

  for (int p = tid; p < WSZ * HDIM; p += 256) {
    int i = p >> 6, d = p & 63;
    int tok = bslide * NTOK + (shuffle ? (i * NWIN + w) : (w * WSZ + i));
    const float* base = qkv + (size_t)tok * (3 * CDIM) + h * HDIM + d;
    qs[p] = base[0] * SCALE;
    ks[p] = base[CDIM];
    vs[p] = base[2 * CDIM];
  }
  if (!shuffle && tid < WSZ) {
    int tok = bslide * NTOK + w * WSZ + tid;
    cx[tid] = coords[tok * 2 + 0];
    cy[tid] = coords[tok * 2 + 1];
  }
  __syncthreads();

  for (int p = tid; p < WSZ * WSZ; p += 256) {
    int i = p / WSZ, j = p - i * WSZ;
    const float* qi = qs + i * HDIM;
    const float* kj = ks + j * HDIM;
    float acc = 0.f;
#pragma unroll 8
    for (int d = 0; d < HDIM; ++d) acc += qi[d] * kj[d];
    if (!shuffle) {
      int idx = abs(pw_idx(cx[i] - cx[j])) + abs(pw_idx(cy[i] - cy[j]));
      acc += rpb[idx * HEADS + h];
    }
    sc[p] = acc;
  }
  __syncthreads();

  if (tid < WSZ) {
    float* row = sc + tid * WSZ;
    float mx = -3.0e38f;
    for (int j = 0; j < WSZ; ++j) mx = fmaxf(mx, row[j]);
    float s = 0.f;
    for (int j = 0; j < WSZ; ++j) { float e = expf(row[j] - mx); row[j] = e; s += e; }
    float inv = 1.f / s;
    for (int j = 0; j < WSZ; ++j) row[j] *= inv;
  }
  __syncthreads();

  for (int p = tid; p < WSZ * HDIM; p += 256) {
    int i = p >> 6, d = p & 63;
    const float* row = sc + i * WSZ;
    float acc = 0.f;
    for (int j = 0; j < WSZ; ++j) acc += row[j] * vs[j * HDIM + d];
    int tok = bslide * NTOK + (shuffle ? (i * NWIN + w) : (w * WSZ + i));
    out[(size_t)tok * CDIM + h * HDIM + d] = acc;
  }
}

// ----------------------------- pooling / heads -----------------------------
__global__ void k_attnnet_logits(const float* __restrict__ t1, const float* __restrict__ w2,
                                 const float* __restrict__ b2, float* __restrict__ logits,
                                 int n) {
  int t = blockIdx.x * 256 + threadIdx.x;
  if (t >= n) return;
  const float* r = t1 + (size_t)t * 256;
  float acc = b2[0];
  for (int j = 0; j < 256; ++j) acc += r[j] * w2[j];
  logits[t] = acc;
}

__global__ void k_softmax_stats(const float* __restrict__ l, int n, float* __restrict__ stats) {
  __shared__ float sm[256];
  int tid = threadIdx.x;
  float mx = -3.0e38f;
  for (int t = tid; t < n; t += 256) mx = fmaxf(mx, l[t]);
  sm[tid] = mx; __syncthreads();
  for (int s = 128; s > 0; s >>= 1) { if (tid < s) sm[tid] = fmaxf(sm[tid], sm[tid + s]); __syncthreads(); }
  mx = sm[0]; __syncthreads();
  float sum = 0.f;
  for (int t = tid; t < n; t += 256) sum += expf(l[t] - mx);
  sm[tid] = sum; __syncthreads();
  for (int s = 128; s > 0; s >>= 1) { if (tid < s) sm[tid] += sm[tid + s]; __syncthreads(); }
  if (tid == 0) { stats[0] = mx; stats[1] = sm[0]; }
}

__global__ void k_pool_feat(const float* __restrict__ xf, const float* __restrict__ logits,
                            const float* __restrict__ stats, float* __restrict__ feat, int n) {
  int c = blockIdx.x * 256 + threadIdx.x;
  if (c >= CDIM) return;
  float mx = stats[0], invZ = 1.f / stats[1];
  float acc = 0.f;
  for (int t = 0; t < n; ++t) acc += expf(logits[t] - mx) * xf[(size_t)t * CDIM + c];
  feat[c] = acc * invZ;
}

__global__ void k_wsi_head(const float* __restrict__ feat, const float* __restrict__ w,
                           const float* __restrict__ b, const float* __restrict__ clin,
                           float* __restrict__ cat) {
  int j = threadIdx.x;  // 256
  float acc = b[j];
  for (int c = 0; c < CDIM; ++c) acc += feat[c] * w[c * 256 + j];
  cat[32 + j] = fmaxf(acc, 0.f);
  if (j < 32) cat[j] = clin[j];
}

__global__ void k_final_head(const float* __restrict__ cat, const float* __restrict__ w1,
                             const float* __restrict__ b1, const float* __restrict__ w2,
                             const float* __restrict__ b2, float* __restrict__ out) {
  __shared__ float h1[128];
  int j = threadIdx.x;  // 128
  float acc = b1[j];
  for (int i = 0; i < 288; ++i) acc += cat[i] * w1[i * 128 + j];
  h1[j] = fmaxf(acc, 0.f);
  __syncthreads();
  if (j < 4) {
    float o = b2[j];
    for (int i = 0; i < 128; ++i) o += h1[i] * w2[i * 4 + j];
    out[j] = o;
  }
}

// ---------------------------------------------------------------------------
extern "C" void kernel_launch(void* const* d_in, const int* in_sizes, int n_in,
                              void* d_out, int out_size, void* d_ws, size_t ws_size,
                              hipStream_t stream) {
  (void)in_sizes; (void)n_in; (void)out_size; (void)ws_size;
  const float* wsi      = (const float*)d_in[0];   // [4, 9800, 1024]
  const int*   coords   = (const int*)  d_in[1];   // [4, 9800, 2]
  const float* clin     = (const float*)d_in[2];   // [32]
  const float* fc1_w    = (const float*)d_in[3];
  const float* fc1_b    = (const float*)d_in[4];
  const float* lnl_g    = (const float*)d_in[5];
  const float* lnl_b    = (const float*)d_in[6];
  const float* w_rpb    = (const float*)d_in[7];   // [15, 8]
  const float* w_qkv_w  = (const float*)d_in[8];
  const float* w_qkv_b  = (const float*)d_in[9];
  const float* w_proj_w = (const float*)d_in[10];
  const float* w_proj_b = (const float*)d_in[11];
  const float* ln1_g    = (const float*)d_in[12];
  const float* ln1_b    = (const float*)d_in[13];
  const float* s_rpb    = (const float*)d_in[14];  // unused (shuffle attn has no bias)
  const float* s_qkv_w  = (const float*)d_in[15];
  const float* s_qkv_b  = (const float*)d_in[16];
  const float* s_proj_w = (const float*)d_in[17];
  const float* s_proj_b = (const float*)d_in[18];
  const float* ln2_g    = (const float*)d_in[19];
  const float* ln2_b    = (const float*)d_in[20];
  const float* mlp_w1   = (const float*)d_in[21];
  const float* mlp_b1   = (const float*)d_in[22];
  const float* mlp_w2   = (const float*)d_in[23];
  const float* mlp_b2   = (const float*)d_in[24];
  const float* ln3_g    = (const float*)d_in[25];
  const float* ln3_b    = (const float*)d_in[26];
  const float* attn_w1  = (const float*)d_in[27];
  const float* attn_b1  = (const float*)d_in[28];
  const float* attn_w2  = (const float*)d_in[29];
  const float* attn_b2  = (const float*)d_in[30];
  const float* wsi_w    = (const float*)d_in[31];
  const float* wsi_b    = (const float*)d_in[32];
  const float* out_w1   = (const float*)d_in[33];
  const float* out_b1   = (const float*)d_in[34];
  const float* out_w2   = (const float*)d_in[35];
  const float* out_b2   = (const float*)d_in[36];
  (void)s_rpb;

  // ----- carve workspace -----
  uintptr_t p = (uintptr_t)d_ws;
  auto carve = [&](size_t bytes) -> void* {
    uintptr_t q = (p + 255) & ~(uintptr_t)255;
    p = q + bytes;
    return (void*)q;
  };
  const size_t Mt = (size_t)MTOK;
  bf16*  Xbf    = (bf16*) carve(Mt * FDIM * 2);
  bf16*  wfc1   = (bf16*) carve((size_t)FDIM * CDIM * 2);          // transposed [N,K]
  bf16*  wqkv1  = (bf16*) carve((size_t)CDIM * 3 * CDIM * 2);
  bf16*  wproj1 = (bf16*) carve((size_t)CDIM * CDIM * 2);
  bf16*  wqkv2  = (bf16*) carve((size_t)CDIM * 3 * CDIM * 2);
  bf16*  wproj2 = (bf16*) carve((size_t)CDIM * CDIM * 2);
  bf16*  wmlp1  = (bf16*) carve((size_t)CDIM * CDIM * 2);
  bf16*  wmlp2  = (bf16*) carve((size_t)CDIM * CDIM * 2);
  bf16*  wat1   = (bf16*) carve((size_t)CDIM * 256 * 2);
  float* bufA   = (float*)carve(Mt * CDIM * 4);
  float* bufB   = (float*)carve(Mt * CDIM * 4);
  float* bufC   = (float*)carve(Mt * CDIM * 4);
  float* bufQ   = (float*)carve(Mt * 3 * CDIM * 4);
  bf16*  actbf  = (bf16*) carve(Mt * CDIM * 2);
  float* smallr = (float*)carve(4096);
  float* logits = bufQ;           // QKV buffer is dead by the time logits are needed
  float* stats  = smallr;         // 2 floats
  float* feat   = smallr + 16;    // 512 floats
  float* cat    = smallr + 544;   // 288 floats

  auto nb = [](size_t n) { return (unsigned)((n + 255) / 256); };
  auto xpose = [&](const float* w, bf16* wt, int K, int N) {
    k_cvt_transpose_bf16<<<dim3(K / 32, N / 32), 256, 0, stream>>>(w, wt, K, N);
  };
  auto gemm = [&](const bf16* A_, const bf16* BT_, const float* bias_, const float* res_,
                  float* C_, int Mg, int Ng, int Kg, int act) {
    dim3 g(Ng / 32, (Mg + 127) / 128), b(256);
    if (res_) {
      if (act == 2)      k_gemm_bf16_wmma<2, true ><<<g, b, 0, stream>>>(A_, BT_, bias_, res_, C_, Mg, Ng, Kg);
      else               k_gemm_bf16_wmma<0, true ><<<g, b, 0, stream>>>(A_, BT_, bias_, res_, C_, Mg, Ng, Kg);
    } else {
      if (act == 1)      k_gemm_bf16_wmma<1, false><<<g, b, 0, stream>>>(A_, BT_, bias_, res_, C_, Mg, Ng, Kg);
      else if (act == 2) k_gemm_bf16_wmma<2, false><<<g, b, 0, stream>>>(A_, BT_, bias_, res_, C_, Mg, Ng, Kg);
      else if (act == 3) k_gemm_bf16_wmma<3, false><<<g, b, 0, stream>>>(A_, BT_, bias_, res_, C_, Mg, Ng, Kg);
      else               k_gemm_bf16_wmma<0, false><<<g, b, 0, stream>>>(A_, BT_, bias_, res_, C_, Mg, Ng, Kg);
    }
  };

  // ----- convert input + transpose weights to bf16 -----
  k_cvt_f32_bf16<<<nb(Mt * FDIM), 256, 0, stream>>>(wsi, Xbf, Mt * FDIM);
  xpose(fc1_w,    wfc1,  FDIM, CDIM);
  xpose(w_qkv_w,  wqkv1, CDIM, 3 * CDIM);
  xpose(w_proj_w, wproj1, CDIM, CDIM);
  xpose(s_qkv_w,  wqkv2, CDIM, 3 * CDIM);
  xpose(s_proj_w, wproj2, CDIM, CDIM);
  xpose(mlp_w1,   wmlp1, CDIM, CDIM);
  xpose(mlp_w2,   wmlp2, CDIM, CDIM);
  xpose(attn_w1,  wat1,  CDIM, 256);

  // ----- fc1 + relu:  h = relu(X @ W1 + b1) -> bufA -----
  gemm(Xbf, wfc1, fc1_b, nullptr, bufA, MTOK, CDIM, FDIM, 1);

  // ----- local (window) attention block: x = gelu(h + Proj(Attn(LN(h)))) ---
  k_ln512<<<MTOK, 256, 0, stream>>>(bufA, lnl_g, lnl_b, nullptr, actbf);
  gemm(actbf, wqkv1, w_qkv_b, nullptr, bufQ, MTOK, 3 * CDIM, CDIM, 0);
  k_window_attn<<<dim3(NWIN, HEADS, SLIDES), 256, 0, stream>>>(bufQ, w_rpb, coords, bufC, 0);
  k_cvt_f32_bf16<<<nb(Mt * CDIM), 256, 0, stream>>>(bufC, actbf, Mt * CDIM);
  gemm(actbf, wproj1, w_proj_b, bufA, bufB, MTOK, CDIM, CDIM, 2);   // gelu(h + proj)

  // ----- shuffle attention block: x = x + Proj(ShuffleAttn(LN(x))) ---------
  k_ln512<<<MTOK, 256, 0, stream>>>(bufB, ln1_g, ln1_b, nullptr, actbf);
  gemm(actbf, wqkv2, s_qkv_b, nullptr, bufQ, MTOK, 3 * CDIM, CDIM, 0);
  k_window_attn<<<dim3(NWIN, HEADS, SLIDES), 256, 0, stream>>>(bufQ, nullptr, coords, bufC, 1);
  k_cvt_f32_bf16<<<nb(Mt * CDIM), 256, 0, stream>>>(bufC, actbf, Mt * CDIM);
  gemm(actbf, wproj2, s_proj_b, bufB, bufA, MTOK, CDIM, CDIM, 0);   // x + proj

  // ----- MLP block: x = x + W2(gelu(W1(LN(x)))) ----------------------------
  k_ln512<<<MTOK, 256, 0, stream>>>(bufA, ln2_g, ln2_b, nullptr, actbf);
  gemm(actbf, wmlp1, mlp_b1, nullptr, bufC, MTOK, CDIM, CDIM, 2);
  k_cvt_f32_bf16<<<nb(Mt * CDIM), 256, 0, stream>>>(bufC, actbf, Mt * CDIM);
  gemm(actbf, wmlp2, mlp_b2, bufA, bufB, MTOK, CDIM, CDIM, 0);

  // ----- final LN, gated pooling, heads ------------------------------------
  k_ln512<<<MTOK, 256, 0, stream>>>(bufB, ln3_g, ln3_b, bufA, actbf);  // xf f32 + bf16
  gemm(actbf, wat1, attn_b1, nullptr, bufC, MTOK, 256, CDIM, 3);       // tanh
  k_attnnet_logits<<<nb(MTOK), 256, 0, stream>>>(bufC, attn_w2, attn_b2, logits, MTOK);
  k_softmax_stats<<<1, 256, 0, stream>>>(logits, MTOK, stats);
  k_pool_feat<<<2, 256, 0, stream>>>(bufA, logits, stats, feat, MTOK);
  k_wsi_head<<<1, 256, 0, stream>>>(feat, wsi_w, wsi_b, clin, cat);
  k_final_head<<<1, 128, 0, stream>>>(cat, out_w1, out_b1, out_w2, out_b2, (float*)d_out);
}
